// VanillaRNN_89240830476471
// MI455X (gfx1250) — matrix operand order repaired and checked by reading, not compiled, not run on hardware
//
#include <hip/hip_runtime.h>
#include <hip/hip_bf16.h>

// ---------------------------------------------------------------------------
// VanillaRNN on MI455X (gfx1250, wave32, WMMA bf16 16x16x32).
//
//  * 512 steps x (1024x1024 @ 1024x512) = ~550 GFLOP, strictly sequential in t.
//  * Whh (bf16, 2MB) is partitioned across 8 M-slice workgroups per batch
//    group; each wave pins its private 16x1024 slice (32KB = 256 VGPRs) in
//    REGISTERS for the entire kernel -> zero per-step weight traffic.
//    (disasm shows A operands at v256+ via s_set_vgpr_msb: register-resident)
//  * LDS holds only the 64KB per-step h staging buffer, filled with
//    GLOBAL_LOAD_ASYNC_TO_LDS_B128 (ASYNCcnt) split in two K-halves so the
//    second half streams from L2 while WMMAs consume the first.
//  * B fragments are double-buffered with distance-2 prefetch so ds_load ->
//    v_wmma latency is off the critical path (avoid s_wait_dscnt 0 stalls).
//  * 8 M-slice WGs per 32-column batch group exchange h through L2 each step:
//    cluster barrier (HW path, NOP when not cluster-dispatched) + agent-scope
//    atomic barrier with release/acquire fences. 16 groups -> 128 WGs.
//  * Layouts: Whh row-major (A frag = two 16B runs/lane), h stored
//    [batch][hidden] so a B column is K-contiguous (one 32B run/lane).
// ---------------------------------------------------------------------------

#define HID   1024
#define SEQ   512
#define BATCH 512
#define NCLS  128
#define NTILE 32     // batch columns per group
#define MROWS 128    // hidden rows per workgroup
#define MSLICES 8    // HID / MROWS
#define BGROUPS 16   // BATCH / NTILE

typedef __attribute__((ext_vector_type(16))) __bf16 v16bf;
typedef __attribute__((ext_vector_type(8)))  __bf16 bf16x8;
typedef __attribute__((ext_vector_type(8)))  float  v8f;

union AFrag { v16bf v; bf16x8 h[2]; };

#define WMMA_BF16(A, B, C) \
  __builtin_amdgcn_wmma_f32_16x16x32_bf16(false, (A), false, (B), (short)0, (C), false, false)

// ---------------------------------------------------------------------------
// Kernel 0: convert weights to bf16, zero h0 and the barrier counters.
// ---------------------------------------------------------------------------
__global__ void __launch_bounds__(256) prep_kernel(
    const float* __restrict__ Whh, const float* __restrict__ Wph,
    __bf16* __restrict__ whh_bf, __bf16* __restrict__ wph_bf,
    __bf16* __restrict__ h0, unsigned* __restrict__ bar)
{
  const unsigned i = blockIdx.x * blockDim.x + threadIdx.x;
  const unsigned stride = gridDim.x * blockDim.x;
  for (unsigned k = i; k < HID * HID; k += stride)   whh_bf[k] = (__bf16)Whh[k];
  for (unsigned k = i; k < NCLS * HID; k += stride)  wph_bf[k] = (__bf16)Wph[k];
  for (unsigned k = i; k < BATCH * HID; k += stride) h0[k] = (__bf16)0.0f;
  if (i < BGROUPS) bar[i] = 0u;
}

// 16-kf compute half with distance-2 double-buffered B prefetch.
template <int KF0>
__device__ __forceinline__ void compute_half(
    const __bf16* __restrict__ bcol0, const __bf16* __restrict__ bcol1,
    int half, const v16bf (&afr)[32], v8f& c0, v8f& c1)
{
  auto ldb = [&](const __bf16* base, int kf) -> v16bf {
    return *(const v16bf*)(base + kf * 32 + half * 16);
  };
  v16bf p00 = ldb(bcol0, KF0 + 0), p10 = ldb(bcol1, KF0 + 0);
  v16bf p01 = ldb(bcol0, KF0 + 1), p11 = ldb(bcol1, KF0 + 1);
#pragma unroll
  for (int i = 0; i < 16; ++i) {
    const int kf = KF0 + i;
    v16bf b0 = (i & 1) ? p01 : p00;
    v16bf b1 = (i & 1) ? p11 : p10;
    c0 = WMMA_BF16(afr[kf], b0, c0);
    c1 = WMMA_BF16(afr[kf], b1, c1);
    if (i + 2 < 16) {               // refill the just-consumed slot, 2 ahead
      if (i & 1) { p01 = ldb(bcol0, kf + 2); p11 = ldb(bcol1, kf + 2); }
      else       { p00 = ldb(bcol0, kf + 2); p10 = ldb(bcol1, kf + 2); }
    }
  }
}

// ---------------------------------------------------------------------------
// Kernel 1: persistent recurrence. grid = MSLICES * BGROUPS = 128 blocks,
// 256 threads (8 waves). Wave w owns rows [mslice*128 + 16w, +16) x 32 cols,
// with its A-slice of Whh held in 256 VGPRs for the whole kernel.
// Dynamic LDS: 64KB h staging only.
// ---------------------------------------------------------------------------
__global__ void __launch_bounds__(256, 1) rnn_kernel(
    const float* __restrict__ x,    // [BATCH][SEQ]
    const float* __restrict__ Whx,  // [HID]
    const float* __restrict__ bh,   // [HID]
    const __bf16* __restrict__ whh, // [HID][HID] bf16 row-major
    __bf16* __restrict__ hA,        // [BATCH][HID] ping
    __bf16* __restrict__ hB,        // [BATCH][HID] pong
    unsigned* __restrict__ bar)     // [BGROUPS]
{
  extern __shared__ char smem_raw[];
  __bf16* Bsh = (__bf16*)smem_raw;   // [NTILE][HID] bf16 = 64KB, LDS offset 0

  const int tid    = threadIdx.x;
  const int mslice = blockIdx.x & (MSLICES - 1);
  const int bgroup = blockIdx.x >> 3;
  const int wave   = tid >> 5;
  const int lane   = tid & 31;
  const int half   = lane >> 4;   // wave32 half (0: lanes 0-15, 1: lanes 16-31)
  const int l15    = lane & 15;

  const int m0 = mslice * MROWS + wave * 16;   // global row base of this wave
  const int rb = m0 + half * 8;                // row of C reg i is rb + i

  // ---- Pin this wave's Whh A-slice (16 rows x 1024 K) in VGPRs: 32 frags,
  //      8 VGPRs each = 256 VGPRs. Read from HBM exactly once. -------------
  v16bf afr[32];
  {
    const __bf16* arow_g = whh + (size_t)(m0 + l15) * HID;
#pragma unroll
    for (int kf = 0; kf < 32; ++kf) {
      AFrag a;
      a.h[0] = *(const bf16x8*)(arow_g + kf * 32 + half * 16);
      a.h[1] = *(const bf16x8*)(arow_g + kf * 32 + half * 16 + 8);
      afr[kf] = a.v;
    }
  }

  // Per-lane epilogue constants: the 8 C-register rows for this lane.
  float whx_r[8], bh_r[8];
#pragma unroll
  for (int i = 0; i < 8; ++i) { whx_r[i] = Whx[rb + i]; bh_r[i] = bh[rb + i]; }

  const int col0 = bgroup * NTILE + l15;       // batch column, N-tile 0
  const int col1 = col0 + 16;                  // batch column, N-tile 1
  const __bf16* bcol0 = Bsh + (size_t)l15 * HID;         // B col tile 0
  const __bf16* bcol1 = Bsh + (size_t)(l15 + 16) * HID;  // B col tile 1

  for (int t = 0; t < SEQ; ++t) {
    __bf16* hin  = (t & 1) ? hB : hA;
    __bf16* hout = (t & 1) ? hA : hB;

    // Issue per-step x reads early; latency hides under the WMMA body.
    const float xb0 = x[(size_t)col0 * SEQ + t];
    const float xb1 = x[(size_t)col1 * SEQ + t];

    // ---- Async-stage B = h[bgroup cols][all K] into LDS (64KB) -----------
    // Layout is identical in global and LDS ([col][K] bf16), so one 32-bit
    // offset serves as both the GVS vaddr and the LDS dest address.
    // Issue both K-halves; wait only for half 0, compute on it while half 1
    // streams from L2 (ASYNCcnt in-order completion).
    {
      const __bf16* src = hin + (size_t)bgroup * NTILE * HID;  // uniform base
#pragma unroll
      for (int it = 0; it < 8; ++it) {                // half 0: K 0..511
        unsigned j = (unsigned)(it * 256 + tid);      // 2048 x 16B chunks
        unsigned off = ((j >> 6) << 11) + ((j & 63u) << 4);
        asm volatile("global_load_async_to_lds_b128 %0, %1, %2"
                     :: "v"(off), "v"(off), "s"(src) : "memory");
      }
#pragma unroll
      for (int it = 0; it < 8; ++it) {                // half 1: K 512..1023
        unsigned j = (unsigned)(it * 256 + tid);
        unsigned off = ((j >> 6) << 11) + ((j & 63u) << 4) + 1024u;
        asm volatile("global_load_async_to_lds_b128 %0, %1, %2"
                     :: "v"(off), "v"(off), "s"(src) : "memory");
      }
      asm volatile("s_wait_asynccnt 8" ::: "memory");  // half 0 landed
    }
    __syncthreads();

    v8f c0 = {}; v8f c1 = {};
    compute_half<0>(bcol0, bcol1, half, afr, c0, c1);   // K 0..511

    asm volatile("s_wait_asynccnt 0" ::: "memory");     // half 1 landed
    __syncthreads();

    compute_half<16>(bcol0, bcol1, half, afr, c0, c1);  // K 512..1023

    // Epilogue: h = tanh(Whh@h + Whx*x_t + bh); pack bf16, store batch-major.
    bf16x8 o0, o1;
#pragma unroll
    for (int i = 0; i < 8; ++i) {
      o0[i] = (__bf16)tanhf(c0[i] + whx_r[i] * xb0 + bh_r[i]);
      o1[i] = (__bf16)tanhf(c1[i] + whx_r[i] * xb1 + bh_r[i]);
    }
    *(bf16x8*)(hout + (size_t)col0 * HID + rb) = o0;   // 8 contiguous rows, 16B
    *(bf16x8*)(hout + (size_t)col1 * HID + rb) = o1;

    // ---- cross-workgroup step barrier (8 M-slice WGs of this batch group) --
    __builtin_amdgcn_fence(__ATOMIC_RELEASE, "agent"); // flush h stores
    __syncthreads();
    __builtin_amdgcn_s_cluster_barrier();              // HW cluster barrier path
    if (tid == 0) {
      __hip_atomic_fetch_add(bar + bgroup, 1u, __ATOMIC_ACQ_REL,
                             __HIP_MEMORY_SCOPE_AGENT);
      const unsigned tgt = (unsigned)MSLICES * (unsigned)(t + 1);
      while (__hip_atomic_load(bar + bgroup, __ATOMIC_ACQUIRE,
                               __HIP_MEMORY_SCOPE_AGENT) < tgt)
        __builtin_amdgcn_s_sleep(2);
    }
    __syncthreads();
    __builtin_amdgcn_fence(__ATOMIC_ACQUIRE, "agent"); // invalidate stale lines
  }
}

// ---------------------------------------------------------------------------
// Kernel 2: p = Wph @ h_last + bp via WMMA bf16. grid = 32 (one 16-col N-tile
// each), 8 waves x 16 rows cover all 128 classes. K streamed from L2 (tiny).
// ---------------------------------------------------------------------------
__global__ void __launch_bounds__(256, 1) proj_kernel(
    const __bf16* __restrict__ wph,   // [NCLS][HID] bf16 row-major
    const __bf16* __restrict__ hlast, // [BATCH][HID] bf16
    const float* __restrict__ bp,     // [NCLS]
    float* __restrict__ p)            // [NCLS][BATCH]
{
  const int tid  = threadIdx.x;
  const int wave = tid >> 5;
  const int lane = tid & 31;
  const int half = lane >> 4;
  const int l15  = lane & 15;
  const int ntile = blockIdx.x;        // 0..31
  const int m0 = wave * 16;

  v8f c = {};
  const __bf16* arow = wph   + (size_t)(m0 + l15) * HID;
  const __bf16* bcol = hlast + (size_t)(ntile * 16 + l15) * HID;
#pragma unroll 8
  for (int kf = 0; kf < HID / 32; ++kf) {
    const int k0 = kf * 32;
    AFrag a;
    a.h[0] = *(const bf16x8*)(arow + k0 + half * 16);
    a.h[1] = *(const bf16x8*)(arow + k0 + half * 16 + 8);
    v16bf b = *(const v16bf*)(bcol + k0 + half * 16);
    c = WMMA_BF16(a.v, b, c);
  }
  const int col = ntile * 16 + l15;
#pragma unroll
  for (int i = 0; i < 8; ++i) {
    const int r = m0 + half * 8 + i;
    p[(size_t)r * BATCH + col] = c[i] + bp[r];
  }
}

// ---------------------------------------------------------------------------
// Kernel 3: softmax over the BATCH axis (faithful to reference), transpose.
// One block per class; wave32 shuffle reductions.
// ---------------------------------------------------------------------------
__global__ void __launch_bounds__(256) softmax_kernel(
    const float* __restrict__ p, float* __restrict__ out)
{
  const int c   = blockIdx.x;      // class
  const int tid = threadIdx.x;
  __shared__ float red[8];
  __shared__ float smax, ssum;

  float v0 = p[(size_t)c * BATCH + tid];
  float v1 = p[(size_t)c * BATCH + tid + 256];

  float m = fmaxf(v0, v1);
  for (int off = 16; off > 0; off >>= 1) m = fmaxf(m, __shfl_xor(m, off, 32));
  if ((tid & 31) == 0) red[tid >> 5] = m;
  __syncthreads();
  if (tid == 0) {
    float mm = red[0];
    for (int i = 1; i < 8; ++i) mm = fmaxf(mm, red[i]);
    smax = mm;
  }
  __syncthreads();

  const float e0 = expf(v0 - smax);
  const float e1 = expf(v1 - smax);
  float s = e0 + e1;
  for (int off = 16; off > 0; off >>= 1) s += __shfl_xor(s, off, 32);
  if ((tid & 31) == 0) red[tid >> 5] = s;
  __syncthreads();
  if (tid == 0) {
    float t = 0.f;
    for (int i = 0; i < 8; ++i) t += red[i];
    ssum = t;
  }
  __syncthreads();

  const float inv = 1.0f / ssum;
  out[(size_t)tid * NCLS + c]         = e0 * inv;
  out[(size_t)(tid + 256) * NCLS + c] = e1 * inv;
}

// ---------------------------------------------------------------------------
// Workspace layout (bytes):
//   whh_bf  : 2,097,152   (1024*1024 bf16)
//   wph_bf  :   262,144   (128*1024 bf16)
//   hA      : 1,048,576   (512*1024 bf16)
//   hB      : 1,048,576
//   p       :   262,144   (128*512 f32)
//   bar     :        64
// ---------------------------------------------------------------------------
#define RNN_LDS_BYTES ((size_t)(NTILE * HID * 2))   // 64KB h staging

extern "C" void kernel_launch(void* const* d_in, const int* in_sizes, int n_in,
                              void* d_out, int out_size, void* d_ws, size_t ws_size,
                              hipStream_t stream) {
  const float* x   = (const float*)d_in[0];
  const float* Whx = (const float*)d_in[1];
  const float* Whh = (const float*)d_in[2];
  const float* Wph = (const float*)d_in[3];
  const float* bh  = (const float*)d_in[4];
  const float* bp  = (const float*)d_in[5];
  float* out = (float*)d_out;

  char* W = (char*)d_ws;
  __bf16*   whh_bf = (__bf16*)(W);
  __bf16*   wph_bf = (__bf16*)(W + 2097152);
  __bf16*   hA     = (__bf16*)(W + 2097152 + 262144);
  __bf16*   hB     = hA + (size_t)BATCH * HID;
  float*    pbuf   = (float*)(W + 2097152 + 262144 + 2 * 1048576);
  unsigned* bar    = (unsigned*)(W + 2097152 + 262144 + 2 * 1048576 + 262144);

  (void)in_sizes; (void)n_in; (void)out_size; (void)ws_size;

  hipFuncSetAttribute((const void*)rnn_kernel,
                      hipFuncAttributeMaxDynamicSharedMemorySize,
                      (int)RNN_LDS_BYTES);

  prep_kernel<<<1024, 256, 0, stream>>>(Whh, Wph, whh_bf, wph_bf, hA, bar);
  rnn_kernel<<<MSLICES * BGROUPS, 256, RNN_LDS_BYTES, stream>>>(
      x, Whx, bh, whh_bf, hA, hB, bar);
  // 512 steps: step t writes (t&1)?hA:hB -> final (t=511) h_last is in hA.
  proj_kernel<<<BATCH / 16, 256, 0, stream>>>(wph_bf, hA, bp, pbuf);
  softmax_kernel<<<NCLS, 256, 0, stream>>>(pbuf, out);
}
